// Message_Passing4OBJ_67095979099058
// MI455X (gfx1250) — matrix operand order repaired and checked by reading, not compiled, not run on hardware
//
#include <hip/hip_runtime.h>

// ---------------- CDNA5 WMMA types ----------------
typedef __attribute__((ext_vector_type(16))) __bf16 v16bf;
typedef __attribute__((ext_vector_type(8)))  float  v8f;

#define D_DIM  4096
#define N_OBJ  1024
#define NPER   64
#define NUMIMG 16
#define RPER   512
#define R_TOT  8192
#define DC     2048   // D/2
#define DT     1024   // D/4

// round-to-nearest-even f32 -> bf16 (bit trick; only used in bandwidth-bound passes)
__device__ __forceinline__ unsigned short f2bf(float f) {
  unsigned int b = __float_as_uint(f);
  b += 0x7FFFu + ((b >> 16) & 1u);
  return (unsigned short)(b >> 16);
}

// elementwise f32 -> bf16, vectorized x4
__global__ __launch_bounds__(256) void cvt_f32_bf16(
    const float* __restrict__ in, unsigned short* __restrict__ out, int n4)
{
  for (int i = blockIdx.x * 256 + threadIdx.x; i < n4; i += gridDim.x * 256) {
    float4 v = ((const float4*)in)[i];
    ushort4 o;
    o.x = f2bf(v.x); o.y = f2bf(v.y); o.z = f2bf(v.z); o.w = f2bf(v.w);
    ((ushort4*)out)[i] = o;
  }
}

// load a 16-element bf16 fragment as two 16B chunks (chunk1 at +secondOfs elements)
union FragU { v16bf v; uint4 q[2]; };
__device__ __forceinline__ v16bf ld_frag(const unsigned short* __restrict__ p, int secondOfs) {
  FragU f;
  f.q[0] = *(const uint4*)(p);
  f.q[1] = *(const uint4*)(p + secondOfs);
  return f.v;
}

// out[M,N] = act( A[M,K] @ W[N,K]^T + bias ) ; act: 0=none 1=relu 2=relu(res+.)
// A, W are bf16. block = 256 threads = 8 waves; block tile 64(M) x 256(N);
// wave tile 32x64 => 8 WMMAs per K-step of 32 against 12 b128 loads.
__global__ __launch_bounds__(256) void gemm_bf16_wmma(
    const unsigned short* __restrict__ A, const unsigned short* __restrict__ W,
    const float* __restrict__ bias, const float* __restrict__ res,
    float* __restrict__ out, int M, int N, int K, int act)
{
  const int lane = threadIdx.x & 31;
  const int wave = threadIdx.x >> 5;
  const int m0 = blockIdx.y * 64 + (wave & 1) * 32;   // two 16-row A frags: m0, m0+16
  const int n0 = blockIdx.x * 256 + (wave >> 1) * 64; // four 16-col B frags
  const int l15 = lane & 15;
  const int hi  = lane >> 4;

  // A frag: lane holds 16 contiguous K starting at k + hi*16
  const unsigned short* arow0 = A + (size_t)(m0 + l15) * K + (hi << 4);
  const unsigned short* arow1 = arow0 + (size_t)16 * K;
  // B frag: lane holds 16 contiguous K starting at k + hi*16 for its column
  const unsigned short* brow = W + (size_t)(n0 + l15) * K + (hi << 4);
  const size_t bstride = (size_t)16 * K;

  v8f acc[2][4] = {};
  for (int k = 0; k < K; k += 32) {
    __builtin_prefetch(arow0 + k + 256, 0, 0);
    __builtin_prefetch(brow + k + 256, 0, 0);
    v16bf a0 = ld_frag(arow0 + k, 8);
    v16bf a1 = ld_frag(arow1 + k, 8);
    v16bf b0 = ld_frag(brow + k, 8);
    v16bf b1 = ld_frag(brow + bstride + k, 8);
    v16bf b2 = ld_frag(brow + 2 * bstride + k, 8);
    v16bf b3 = ld_frag(brow + 3 * bstride + k, 8);
    acc[0][0] = __builtin_amdgcn_wmma_f32_16x16x32_bf16(false, a0, false, b0, (short)0, acc[0][0], false, false);
    acc[0][1] = __builtin_amdgcn_wmma_f32_16x16x32_bf16(false, a0, false, b1, (short)0, acc[0][1], false, false);
    acc[0][2] = __builtin_amdgcn_wmma_f32_16x16x32_bf16(false, a0, false, b2, (short)0, acc[0][2], false, false);
    acc[0][3] = __builtin_amdgcn_wmma_f32_16x16x32_bf16(false, a0, false, b3, (short)0, acc[0][3], false, false);
    acc[1][0] = __builtin_amdgcn_wmma_f32_16x16x32_bf16(false, a1, false, b0, (short)0, acc[1][0], false, false);
    acc[1][1] = __builtin_amdgcn_wmma_f32_16x16x32_bf16(false, a1, false, b1, (short)0, acc[1][1], false, false);
    acc[1][2] = __builtin_amdgcn_wmma_f32_16x16x32_bf16(false, a1, false, b2, (short)0, acc[1][2], false, false);
    acc[1][3] = __builtin_amdgcn_wmma_f32_16x16x32_bf16(false, a1, false, b3, (short)0, acc[1][3], false, false);
  }

  // C/D layout: VGPR r -> row (r | hi*8), lane&15 -> column
  #pragma unroll
  for (int mi = 0; mi < 2; ++mi) {
    const int rbase = m0 + mi * 16 + (hi << 3);
    #pragma unroll
    for (int nf = 0; nf < 4; ++nf) {
      const int col = n0 + nf * 16 + l15;
      const float bv = bias[col];
      #pragma unroll
      for (int r = 0; r < 8; ++r) {
        const int row = rbase + r;
        float v = acc[mi][nf][r] + bv;
        if (act == 1)      v = fmaxf(v, 0.0f);
        else if (act == 2) v = fmaxf(v + res[(size_t)row * N + col], 0.0f);
        out[(size_t)row * N + col] = v;
      }
    }
  }
}

// atten[r] = sum_d ws[subj,d]*wo[obj,d]*phr[r,d]*Ww_w[d] + Ww_b
__global__ __launch_bounds__(256) void atten_kernel(
    const float* __restrict__ phr, const float* __restrict__ wsA,
    const float* __restrict__ woA, const int* __restrict__ rel,
    const float* __restrict__ Www, const float* __restrict__ Wwb,
    float* __restrict__ atten)
{
  const int r = blockIdx.x;
  const int s = rel[3 * r + 1];
  const int o = rel[3 * r + 2];
  const float4* pr = (const float4*)(phr + (size_t)r * D_DIM);
  const float4* ps = (const float4*)(wsA + (size_t)s * D_DIM);
  const float4* po = (const float4*)(woA + (size_t)o * D_DIM);
  const float4* pw = (const float4*)Www;
  float acc = 0.f;
  for (int i = threadIdx.x; i < D_DIM / 4; i += 256) {
    float4 a = ps[i], b = po[i], c = pr[i], w = pw[i];
    acc += a.x * b.x * c.x * w.x + a.y * b.y * c.y * w.y +
           a.z * b.z * c.z * w.z + a.w * b.w * c.w * w.w;
  }
  #pragma unroll
  for (int off = 16; off > 0; off >>= 1) acc += __shfl_down(acc, off, 32);
  __shared__ float red[8];
  if ((threadIdx.x & 31) == 0) red[threadIdx.x >> 5] = acc;
  __syncthreads();
  if (threadIdx.x == 0) {
    float t = 0.f;
    #pragma unroll
    for (int i = 0; i < 8; ++i) t += red[i];
    atten[r] = t + Wwb[0];
  }
}

// per-image: LDS scatter -> sigmoid -> self-mask -> normalize (by row-sum of the
// *second* index, matching the reference broadcast) -> ctx = [A@conv ; (A^T)@conv]
// ctx is written in bf16 (feeds the Wt1 WMMA GEMM).
__global__ __launch_bounds__(256) void image_kernel(
    const int* __restrict__ rel, const float* __restrict__ atten,
    const float* __restrict__ conv, unsigned short* __restrict__ ctx)
{
  const int img = blockIdx.x;
  const int tid = threadIdx.x;
  __shared__ float Alds[NPER * NPER];
  __shared__ float convt[NPER * 64];
  __shared__ float Sinv[NPER];
  __shared__ int headsh;
  if (tid == 0) headsh = 0x7fffffff;
  for (int e = tid; e < NPER * NPER; e += 256) Alds[e] = 0.f;
  __syncthreads();

  int hmin = 0x7fffffff;
  for (int e = tid; e < 2 * RPER; e += 256) {
    int rr = img * RPER + (e >> 1);
    int idx = rel[3 * rr + 1 + (e & 1)];
    hmin = min(hmin, idx);
  }
  atomicMin(&headsh, hmin);
  __syncthreads();
  const int head = headsh;

  for (int e = tid; e < RPER; e += 256) {
    int rr = img * RPER + e;
    int si = rel[3 * rr + 1] - head;
    int oi = rel[3 * rr + 2] - head;
    atomicAdd(&Alds[si * NPER + oi], atten[rr]);
  }
  __syncthreads();

  for (int e = tid; e < NPER * NPER; e += 256) {
    float v = Alds[e];
    v = 1.0f / (1.0f + __expf(-v));
    if ((e >> 6) == (e & 63)) v = 0.f;   // self mask
    Alds[e] = v;
  }
  __syncthreads();

  if (tid < NPER) {
    float s = 0.f;
    for (int kk = 0; kk < NPER; ++kk) s += Alds[tid * NPER + kk];
    Sinv[tid] = 1.0f / s;
  }
  __syncthreads();

  const int x  = tid & 63;
  const int ib = (tid >> 6) << 4;
  for (int dt = 0; dt < DC; dt += 64) {
    for (int e = tid; e < NPER * 64; e += 256) {
      int j = e >> 6, xx = e & 63;
      convt[e] = conv[(size_t)(img * NPER + j) * DC + dt + xx];
    }
    __syncthreads();
    for (int e = 0; e < 16; ++e) {
      int i = ib + e;
      float s0 = 0.f, s1 = 0.f;
      for (int j = 0; j < NPER; ++j) {
        float cv = convt[j * 64 + x];
        s0 += Alds[i * NPER + j] * Sinv[j] * cv;  // A[i,j]/S[j]
        s1 += Alds[j * NPER + i] * cv;            // A[j,i] * 1/S[i] (scaled after)
      }
      s1 *= Sinv[i];
      unsigned short* dst = ctx + (size_t)(img * NPER + i) * D_DIM;
      dst[dt + x]      = f2bf(s0);
      dst[DC + dt + x] = f2bf(s1);
    }
    __syncthreads();
  }
}

// h2 = relu( LN(h) * g + b ), one block per row of 1024, bf16 output
__global__ __launch_bounds__(256) void ln_kernel(
    const float* __restrict__ h, const float* __restrict__ g,
    const float* __restrict__ b, unsigned short* __restrict__ h2)
{
  const int row = blockIdx.x;
  const int tid = threadIdx.x;
  float4 v = ((const float4*)(h + (size_t)row * DT))[tid];
  float s  = v.x + v.y + v.z + v.w;
  float s2 = v.x * v.x + v.y * v.y + v.z * v.z + v.w * v.w;
  #pragma unroll
  for (int off = 16; off > 0; off >>= 1) {
    s  += __shfl_down(s, off, 32);
    s2 += __shfl_down(s2, off, 32);
  }
  __shared__ float rs[8], rq[8];
  __shared__ float mu_s, rstd_s;
  if ((tid & 31) == 0) { rs[tid >> 5] = s; rq[tid >> 5] = s2; }
  __syncthreads();
  if (tid == 0) {
    float a = 0.f, c = 0.f;
    #pragma unroll
    for (int i = 0; i < 8; ++i) { a += rs[i]; c += rq[i]; }
    float mu  = a / (float)DT;
    float var = c / (float)DT - mu * mu;
    mu_s = mu;
    rstd_s = rsqrtf(var + 1e-5f);
  }
  __syncthreads();
  const float mu = mu_s, rstd = rstd_s;
  float4 gv = ((const float4*)g)[tid];
  float4 bv = ((const float4*)b)[tid];
  ushort4 o;
  o.x = f2bf(fmaxf((v.x - mu) * rstd * gv.x + bv.x, 0.f));
  o.y = f2bf(fmaxf((v.y - mu) * rstd * gv.y + bv.y, 0.f));
  o.z = f2bf(fmaxf((v.z - mu) * rstd * gv.z + bv.z, 0.f));
  o.w = f2bf(fmaxf((v.w - mu) * rstd * gv.w + bv.w, 0.f));
  ((ushort4*)(h2 + (size_t)row * DT))[tid] = o;
}

extern "C" void kernel_launch(void* const* d_in, const int* in_sizes, int n_in,
                              void* d_out, int out_size, void* d_ws, size_t ws_size,
                              hipStream_t stream) {
  const float* obj   = (const float*)d_in[0];
  const float* phr   = (const float*)d_in[1];
  const int*   rel   = (const int*)d_in[3];
  const float* Ws_w  = (const float*)d_in[5];
  const float* Ws_b  = (const float*)d_in[6];
  const float* Wo_w  = (const float*)d_in[7];
  const float* Wo_b  = (const float*)d_in[8];
  const float* Ww_w  = (const float*)d_in[9];
  const float* Ww_b  = (const float*)d_in[10];
  const float* Wc_w  = (const float*)d_in[11];
  const float* Wc_b  = (const float*)d_in[12];
  const float* Wt1_w = (const float*)d_in[13];
  const float* Wt1_b = (const float*)d_in[14];
  const float* ln_g  = (const float*)d_in[15];
  const float* ln_b  = (const float*)d_in[16];
  const float* Wt2_w = (const float*)d_in[17];
  const float* Wt2_b = (const float*)d_in[18];
  float* out = (float*)d_out;

  // ---- workspace layout ----
  char* base = (char*)d_ws;
  float* ws_all = (float*)base;                 base += (size_t)N_OBJ * D_DIM * 4;  // 16MB
  float* wo_all = (float*)base;                 base += (size_t)N_OBJ * D_DIM * 4;  // 16MB
  float* conv   = (float*)base;                 base += (size_t)N_OBJ * DC * 4;     // 8MB
  float* hbuf   = (float*)base;                 base += (size_t)N_OBJ * DT * 4;     // 4MB
  float* atten  = (float*)base;                 base += (size_t)R_TOT * 4;          // 32KB
  unsigned short* obj_bf = (unsigned short*)base; base += (size_t)N_OBJ * D_DIM * 2; // 8MB
  unsigned short* Ws_bf  = (unsigned short*)base; base += (size_t)D_DIM * D_DIM * 2; // 32MB
  unsigned short* Wo_bf  = (unsigned short*)base; base += (size_t)D_DIM * D_DIM * 2; // 32MB
  unsigned short* Wc_bf  = (unsigned short*)base; base += (size_t)DC * D_DIM * 2;    // 16MB
  unsigned short* Wt1_bf = (unsigned short*)base; base += (size_t)DT * D_DIM * 2;    // 8MB
  unsigned short* Wt2_bf = (unsigned short*)base; base += (size_t)D_DIM * DT * 2;    // 8MB
  unsigned short* ctx_bf = (unsigned short*)base; base += (size_t)N_OBJ * D_DIM * 2; // 8MB
  unsigned short* h2_bf  = (unsigned short*)base; base += (size_t)N_OBJ * DT * 2;    // 2MB

  dim3 blk(256);
  // ---- one-time f32 -> bf16 conversions (bandwidth bound) ----
  {
    int n4;
    n4 = (N_OBJ * D_DIM) / 4; cvt_f32_bf16<<<dim3((n4 + 255) / 256), blk, 0, stream>>>(obj,   obj_bf, n4);
    n4 = (D_DIM * D_DIM) / 4; cvt_f32_bf16<<<dim3((n4 + 255) / 256), blk, 0, stream>>>(Ws_w,  Ws_bf,  n4);
    n4 = (D_DIM * D_DIM) / 4; cvt_f32_bf16<<<dim3((n4 + 255) / 256), blk, 0, stream>>>(Wo_w,  Wo_bf,  n4);
    n4 = (DC * D_DIM)    / 4; cvt_f32_bf16<<<dim3((n4 + 255) / 256), blk, 0, stream>>>(Wc_w,  Wc_bf,  n4);
    n4 = (DT * D_DIM)    / 4; cvt_f32_bf16<<<dim3((n4 + 255) / 256), blk, 0, stream>>>(Wt1_w, Wt1_bf, n4);
    n4 = (D_DIM * DT)    / 4; cvt_f32_bf16<<<dim3((n4 + 255) / 256), blk, 0, stream>>>(Wt2_w, Wt2_bf, n4);
  }

  // ---- pipeline ----
  gemm_bf16_wmma<<<dim3(D_DIM / 256, N_OBJ / 64), blk, 0, stream>>>(
      obj_bf, Ws_bf, Ws_b, nullptr, ws_all, N_OBJ, D_DIM, D_DIM, 0);
  gemm_bf16_wmma<<<dim3(D_DIM / 256, N_OBJ / 64), blk, 0, stream>>>(
      obj_bf, Wo_bf, Wo_b, nullptr, wo_all, N_OBJ, D_DIM, D_DIM, 0);
  gemm_bf16_wmma<<<dim3(DC / 256, N_OBJ / 64), blk, 0, stream>>>(
      obj_bf, Wc_bf, Wc_b, nullptr, conv, N_OBJ, DC, D_DIM, 1);
  atten_kernel<<<dim3(R_TOT), blk, 0, stream>>>(phr, ws_all, wo_all, rel, Ww_w, Ww_b, atten);
  image_kernel<<<dim3(NUMIMG), blk, 0, stream>>>(rel, atten, conv, ctx_bf);
  gemm_bf16_wmma<<<dim3(DT / 256, N_OBJ / 64), blk, 0, stream>>>(
      ctx_bf, Wt1_bf, Wt1_b, nullptr, hbuf, N_OBJ, DT, D_DIM, 0);
  ln_kernel<<<dim3(N_OBJ), blk, 0, stream>>>(hbuf, ln_g, ln_b, h2_bf);
  gemm_bf16_wmma<<<dim3(D_DIM / 256, N_OBJ / 64), blk, 0, stream>>>(
      h2_bf, Wt2_bf, Wt2_b, obj, out, N_OBJ, D_DIM, DT, 2);
}
